// GRU_20100446945320
// MI455X (gfx1250) — matrix operand order repaired and checked by reading, not compiled, not run on hardware
//
#include <hip/hip_runtime.h>
#include <hip/hip_bf16.h>

// GRU with per-step LayerNorm, B=128, T=256, D=256, U=512.
// Phase 0: convert + tile W/U weights to bf16 WMMA B-fragment layout (ws).
// Phase 1: parallel input projections (gz,gr,gh) via v_wmma_f32_16x16x32_bf16.
// Phase 2: 8 persistent workgroups (16 batch rows each) run the 256-step scan
//          with h in LDS; no cross-workgroup sync needed (batch-parallel scan).
//
// Workspace layout (bytes):
//   Ut (3 x 512x512 bf16 tiled)  : 3 * 512 KB
//   Wt (3 x 256x512 bf16 tiled)  : 3 * 256 KB
//   gates (3 x 32768x512 bf16)   : 3 * 32  MB
//   total ~= 98.3 MB

typedef __attribute__((ext_vector_type(16))) __bf16 v16bf;
typedef __attribute__((ext_vector_type(8)))  float  v8f;

#define LN_EPS 1e-3f

__device__ __forceinline__ float sigmoid_f(float x) {
  return 1.0f / (1.0f + __expf(-x));
}
__device__ __forceinline__ float tanh_f(float x) {
  // tanh(x) = 1 - 2/(exp(2x)+1)  (saturates correctly for |x| large)
  return 1.0f - 2.0f / (__expf(2.0f * x) + 1.0f);
}

// Load A fragment (16x32 bf16, ISA layout) from an LDS row-major bf16 buffer.
// Lane l: row = l&15, half = l>>4. VGPR j holds K = (j>>2)*16 + half*8 + (j&3)*2 + {0,1}.
__device__ __forceinline__ v16bf a_frag_lds(const __bf16* buf, int stride,
                                            int row, int half, int kbase) {
  v16bf a;
#pragma unroll
  for (int j = 0; j < 8; ++j) {
    int K = kbase + ((j >> 2) << 4) + (half << 3) + ((j & 3) << 1);
    a[2 * j]     = buf[row * stride + K];
    a[2 * j + 1] = buf[row * stride + K + 1];
  }
  return a;
}

// ---------------------------------------------------------------------------
// Kernel 0: convert fp32 weight [K x 512] to bf16, tiled as [K/32][32][32][16]
// so a WMMA B-fragment (32x16) is 1 KB contiguous: lane l loads 16 bf16 at
// (((kb*32 + nb)*32 + l)*16).
// ---------------------------------------------------------------------------
__global__ void tile_cvt(const float* __restrict__ src, __bf16* __restrict__ dst,
                         int K) {
  int i = blockIdx.x * 256 + threadIdx.x;
  if (i >= K * 512) return;
  int k = i >> 9, n = i & 511;
  int kb = k >> 5, nb = n >> 4;
  int idx = (((kb * 32) + nb) * 32 + (((k >> 3) & 1) << 4) + (n & 15)) * 16 + (k & 7) + (((k >> 4) & 1) << 3);
  dst[idx] = (__bf16)src[i];
}

// ---------------------------------------------------------------------------
// Kernel 1: gates[m, n] = bf16( x[m,:] @ W[:,n] + bias[n] ),  m in [0,32768)
// One block = one 16-row m-tile, 8 waves cover 32 n-tiles (4 each), K=256.
// ---------------------------------------------------------------------------
__global__ void gate_gemm(const float* __restrict__ x,
                          const __bf16* __restrict__ Wt,
                          const float* __restrict__ bias,
                          __bf16* __restrict__ gout) {
  __shared__ __bf16 xs[16 * 256];  // 8 KB staged bf16 A tile
  const int tid = threadIdx.x;
  const int m0 = blockIdx.x * 16;

  for (int i = tid; i < 16 * 256; i += 256) {
    int r = i >> 8, k = i & 255;
    xs[i] = (__bf16)x[(m0 + r) * 256 + k];
  }
  __syncthreads();

  const int wave = tid >> 5, lane = tid & 31;
  const int half = lane >> 4, col = lane & 15, row = lane & 15;
  const v8f vzero = {0.f, 0.f, 0.f, 0.f, 0.f, 0.f, 0.f, 0.f};

  v8f acc[4];
#pragma unroll
  for (int q = 0; q < 4; ++q) acc[q] = vzero;

  for (int kb = 0; kb < 8; ++kb) {
    v16bf a = a_frag_lds(xs, 256, row, half, kb * 32);
#pragma unroll
    for (int q = 0; q < 4; ++q) {
      int nb = wave * 4 + q;
      v16bf b = *(const v16bf*)(Wt + (size_t)((kb * 32 + nb) * 32 + lane) * 16);
      acc[q] = __builtin_amdgcn_wmma_f32_16x16x32_bf16(
          false, a, false, b, (short)0, acc[q], false, false);
    }
  }

#pragma unroll
  for (int q = 0; q < 4; ++q) {
    int n = (wave * 4 + q) * 16 + col;
    float bv = bias[n];
#pragma unroll
    for (int i = 0; i < 8; ++i) {
      int m = m0 + i + 8 * half;
      gout[(size_t)m * 512 + n] = (__bf16)(acc[q][i] + bv);
    }
  }
}

// ---------------------------------------------------------------------------
// Kernel 2: the scan. Grid = 8 blocks (16 batch rows each), 512 threads
// = 16 waves. Wave w owns n-tiles {2w, 2w+1} and LayerNorm row w.
// LDS: h fp32 (32K) + h bf16 shadow (16K) + r*h bf16 (16K) = 64 KB.
// ---------------------------------------------------------------------------
__global__ void gru_scan(const __bf16* __restrict__ Gz,
                         const __bf16* __restrict__ Gr,
                         const __bf16* __restrict__ Gh,
                         const __bf16* __restrict__ Uzt,
                         const __bf16* __restrict__ Urt,
                         const __bf16* __restrict__ Uht,
                         const float* __restrict__ gamma,
                         const float* __restrict__ beta,
                         float* __restrict__ out) {
  __shared__ float  h32[16 * 512];   // fp32 hidden state
  __shared__ __bf16 hbf[16 * 512];   // bf16 shadow (WMMA A operand)
  __shared__ __bf16 rhbf[16 * 512];  // r * h_prev (WMMA A operand)

  const int b0 = blockIdx.x * 16;
  const int tid = threadIdx.x, wave = tid >> 5, lane = tid & 31;
  const int half = lane >> 4, col = lane & 15, row = lane & 15;
  const int nb0 = wave * 2;
  const v8f vzero = {0.f, 0.f, 0.f, 0.f, 0.f, 0.f, 0.f, 0.f};

  for (int i = tid; i < 16 * 512; i += 512) {
    h32[i] = 0.0f;
    hbf[i] = (__bf16)0.0f;
  }
  __syncthreads();

  for (int t = 0; t < 256; ++t) {
    // ---- z and r gates: acc = h_prev @ U_z / U_r --------------------------
    v8f az[2], ar[2];
    az[0] = az[1] = ar[0] = ar[1] = vzero;
    for (int kb = 0; kb < 16; ++kb) {
      v16bf a = a_frag_lds(hbf, 512, row, half, kb * 32);
#pragma unroll
      for (int q = 0; q < 2; ++q) {
        int nb = nb0 + q;
        v16bf bz = *(const v16bf*)(Uzt + (size_t)((kb * 32 + nb) * 32 + lane) * 16);
        v16bf br = *(const v16bf*)(Urt + (size_t)((kb * 32 + nb) * 32 + lane) * 16);
        az[q] = __builtin_amdgcn_wmma_f32_16x16x32_bf16(
            false, a, false, bz, (short)0, az[q], false, false);
        ar[q] = __builtin_amdgcn_wmma_f32_16x16x32_bf16(
            false, a, false, br, (short)0, ar[q], false, false);
      }
    }

    float zv[2][8];  // keep z for the gated update
#pragma unroll
    for (int q = 0; q < 2; ++q) {
      int n = (nb0 + q) * 16 + col;
#pragma unroll
      for (int i = 0; i < 8; ++i) {
        int m = i + 8 * half;
        int grow = (b0 + m) * 256 + t;  // row in gate tensors [B*T, U]
        float z = sigmoid_f((float)Gz[(size_t)grow * 512 + n] + az[q][i]);
        float r = sigmoid_f((float)Gr[(size_t)grow * 512 + n] + ar[q][i]);
        zv[q][i] = z;
        rhbf[m * 512 + n] = (__bf16)(r * h32[m * 512 + n]);
      }
    }
    __syncthreads();

    // ---- candidate: acc = (r*h_prev) @ U_h --------------------------------
    v8f ah[2];
    ah[0] = ah[1] = vzero;
    for (int kb = 0; kb < 16; ++kb) {
      v16bf a = a_frag_lds(rhbf, 512, row, half, kb * 32);
#pragma unroll
      for (int q = 0; q < 2; ++q) {
        int nb = nb0 + q;
        v16bf bh = *(const v16bf*)(Uht + (size_t)((kb * 32 + nb) * 32 + lane) * 16);
        ah[q] = __builtin_amdgcn_wmma_f32_16x16x32_bf16(
            false, a, false, bh, (short)0, ah[q], false, false);
      }
    }

    // ---- gated update (owner lanes write their own h cells) --------------
#pragma unroll
    for (int q = 0; q < 2; ++q) {
      int n = (nb0 + q) * 16 + col;
#pragma unroll
      for (int i = 0; i < 8; ++i) {
        int m = i + 8 * half;
        int grow = (b0 + m) * 256 + t;
        float hh = tanh_f((float)Gh[(size_t)grow * 512 + n] + ah[q][i]);
        float hp = h32[m * 512 + n];
        float z = zv[q][i];
        h32[m * 512 + n] = (1.0f - z) * hh + z * hp;
      }
    }
    __syncthreads();

    // ---- LayerNorm: wave w normalizes row w over U=512 --------------------
    {
      const int r = wave;
      float s = 0.0f, ss = 0.0f;
#pragma unroll
      for (int i = 0; i < 16; ++i) {
        float v = h32[r * 512 + lane + 32 * i];
        s += v;
        ss += v * v;
      }
#pragma unroll
      for (int off = 16; off >= 1; off >>= 1) {
        s += __shfl_xor(s, off, 32);
        ss += __shfl_xor(ss, off, 32);
      }
      float mean = s * (1.0f / 512.0f);
      float var = ss * (1.0f / 512.0f) - mean * mean;
      float rsig = rsqrtf(var + LN_EPS);
#pragma unroll
      for (int i = 0; i < 16; ++i) {
        int n = lane + 32 * i;
        float v = (h32[r * 512 + n] - mean) * rsig * gamma[n] + beta[n];
        h32[r * 512 + n] = v;
        hbf[r * 512 + n] = (__bf16)v;
      }
    }
    __syncthreads();
  }

  // final hidden state -> out[b0:b0+16, :]  (contiguous slice)
  for (int i = tid; i < 16 * 512; i += 512) out[(size_t)b0 * 512 + i] = h32[i];
}

// ---------------------------------------------------------------------------
extern "C" void kernel_launch(void* const* d_in, const int* in_sizes, int n_in,
                              void* d_out, int out_size, void* d_ws, size_t ws_size,
                              hipStream_t stream) {
  const float* x     = (const float*)d_in[0];
  const float* W_z   = (const float*)d_in[1];
  const float* U_z   = (const float*)d_in[2];
  const float* b_z   = (const float*)d_in[3];
  const float* W_r   = (const float*)d_in[4];
  const float* U_r   = (const float*)d_in[5];
  const float* b_r   = (const float*)d_in[6];
  const float* W_h   = (const float*)d_in[7];
  const float* U_h   = (const float*)d_in[8];
  const float* b_h   = (const float*)d_in[9];
  const float* gamma = (const float*)d_in[10];
  const float* beta  = (const float*)d_in[11];

  __bf16* Uzt = (__bf16*)d_ws;
  __bf16* Urt = Uzt + (size_t)512 * 512;
  __bf16* Uht = Urt + (size_t)512 * 512;
  __bf16* Wzt = Uht + (size_t)512 * 512;
  __bf16* Wrt = Wzt + (size_t)256 * 512;
  __bf16* Wht = Wrt + (size_t)256 * 512;
  __bf16* Gz  = Wht + (size_t)256 * 512;
  __bf16* Gr  = Gz + (size_t)32768 * 512;
  __bf16* Gh  = Gr + (size_t)32768 * 512;

  // Phase 0: weight convert + tile (bf16, WMMA B-fragment blocked layout).
  const int nU = 512 * 512, nW = 256 * 512;
  tile_cvt<<<(nU + 255) / 256, 256, 0, stream>>>(U_z, Uzt, 512);
  tile_cvt<<<(nU + 255) / 256, 256, 0, stream>>>(U_r, Urt, 512);
  tile_cvt<<<(nU + 255) / 256, 256, 0, stream>>>(U_h, Uht, 512);
  tile_cvt<<<(nW + 255) / 256, 256, 0, stream>>>(W_z, Wzt, 256);
  tile_cvt<<<(nW + 255) / 256, 256, 0, stream>>>(W_r, Wrt, 256);
  tile_cvt<<<(nW + 255) / 256, 256, 0, stream>>>(W_h, Wht, 256);

  // Phase 1: batched input projections for all timesteps (full-chip WMMA).
  gate_gemm<<<2048, 256, 0, stream>>>(x, Wzt, b_z, Gz);
  gate_gemm<<<2048, 256, 0, stream>>>(x, Wrt, b_r, Gr);
  gate_gemm<<<2048, 256, 0, stream>>>(x, Wht, b_h, Gh);

  // Phase 2: batch-parallel sequential scan (8 independent workgroups).
  gru_scan<<<8, 512, 0, stream>>>(Gz, Gr, Gh, Uzt, Urt, Uht, gamma, beta,
                                  (float*)d_out);
}
